// DecoderPythonCRF_7567732375884
// MI455X (gfx1250) — matrix operand order repaired and checked by reading, not compiled, not run on hardware
//
#include <hip/hip_runtime.h>
#include <hip/hip_bf16.h>

// ---- problem dims (fixed by reference) ----
#define DB   128      // batch
#define DT   256      // decode steps
#define DHH  512      // hidden H
#define DE   256      // embed E
#define DK   256      // tags K
#define TENC 256      // encoder length
#define D2H  1024     // 2*H

typedef __attribute__((ext_vector_type(16))) __bf16 v16bf;
typedef __attribute__((ext_vector_type(4)))  __bf16 v4bf;
typedef __attribute__((ext_vector_type(8)))  float  v8f;

__device__ __forceinline__ float sigmoidf_fast(float x) {
    return 1.0f / (1.0f + __expf(-x));
}

// ============================================================
// Prep kernels (run every call: deterministic, no caching)
// ============================================================

// (T,B,2H) fp32 -> (B,T,2H) bf16   (L2-resident working copy: 67 MB < 192 MB L2)
__global__ void convert_enc_kernel(const float* __restrict__ in, __bf16* __restrict__ out) {
    size_t o = (size_t)blockIdx.x * blockDim.x + threadIdx.x;   // over B*T*2H
    int c  = (int)(o % D2H);
    int bt = (int)(o / D2H);
    int t  = bt % TENC;
    int b  = bt / TENC;
    out[o] = (__bf16)in[((size_t)t * DB + b) * D2H + c];
}

// Pack W (N x Kt row-major, fp32) into WMMA B-matrix (Kt x N) bf16 fragments.
// B 32x16 bf16 layout: lane l -> column n = l%16; k_local = (slot/8)*16 + (l/16)*8 + slot%8.
// Tile order: tile(tn, tk) at ((tn*KT + tk)*32 + lane)*16 + slot.
__global__ void pack_b_kernel(const float* __restrict__ W, __bf16* __restrict__ out,
                              int N, int Kt) {
    size_t o = (size_t)blockIdx.x * blockDim.x + threadIdx.x;   // over N*Kt
    int s    = (int)(o & 15);
    int l    = (int)((o >> 4) & 31);
    int tile = (int)(o >> 9);
    int KT   = Kt >> 5;
    int tk   = tile % KT;
    int tn   = tile / KT;
    int n    = tn * 16 + (l & 15);
    int k    = tk * 32 + ((s >> 3) << 4) + ((l >> 4) << 3) + (s & 7);
    out[o] = (__bf16)W[(size_t)n * Kt + k];
}

__global__ void zero_kernel(float* p, int n) {
    int i = blockIdx.x * blockDim.x + threadIdx.x;
    if (i < n) p[i] = 0.0f;
}

// ============================================================
// A-fragment builder: 16x32 bf16 A tile from fp32 row(s).
// A layout: lane l -> row m = l%16; VGPR v: k = (v/4)*16 + (l/16)*8 + (v%4)*2 + half
// => slot s: k_local = (s/8)*16 + (l/16)*8 + (s%8); pairs are K-contiguous.
// ============================================================
__device__ __forceinline__ v16bf build_a_frag(const float* __restrict__ Xrow0, int K0,
                                              const float* __restrict__ Xrow1,
                                              int kt0, int khalf) {
    v16bf a;
#pragma unroll
    for (int p = 0; p < 8; ++p) {
        int s = 2 * p;
        int k = kt0 + ((s >> 3) << 4) + khalf + (s & 7);
        float f0, f1;
        if (k < K0) { f0 = Xrow0[k]; f1 = Xrow0[k + 1]; }
        else        { f0 = Xrow1[k - K0]; f1 = Xrow1[k - K0 + 1]; }
        a[s]     = (__bf16)f0;
        a[s + 1] = (__bf16)f1;
    }
    return a;
}

// ============================================================
// Generic bf16 WMMA GEMM:  Y = act( [X0|X1](MxKtot) @ Bp(KtotxN) + bias )
// Each wave: one 16(M) x 64(N) strip, full K loop. act: 0=none, 1=relu.
// ============================================================
__global__ __launch_bounds__(256)
void gemm_bf16_kernel(const float* __restrict__ X0, int K0,
                      const float* __restrict__ X1,
                      const __bf16* __restrict__ Bp,
                      const float* __restrict__ bias,
                      float* __restrict__ Y,
                      int M, int N, int Ktot, int act) {
    int wave = blockIdx.x * (blockDim.x >> 5) + (threadIdx.x >> 5);
    int lane = threadIdx.x & 31;
    int ngroups = N >> 6;
    int KT = Ktot >> 5;
    if (wave >= (M >> 4) * ngroups) return;       // wave-uniform: EXEC stays all-ones
    int mtile = wave / ngroups;
    int ng    = wave % ngroups;

    v8f acc0 = {}, acc1 = {}, acc2 = {}, acc3 = {};

    int m = mtile * 16 + (lane & 15);
    int khalf = (lane >> 4) << 3;
    const float* Xrow0 = X0 + (size_t)m * K0;
    const float* Xrow1 = X1 ? X1 + (size_t)m * (Ktot - K0) : (const float*)0;

    for (int tk = 0; tk < KT; ++tk) {
        v16bf a = build_a_frag(Xrow0, K0, Xrow1, tk * 32, khalf);
        const __bf16* bp0 = Bp + ((size_t)((ng * 4 + 0) * KT + tk) * 32 + lane) * 16;
        const __bf16* bp1 = Bp + ((size_t)((ng * 4 + 1) * KT + tk) * 32 + lane) * 16;
        const __bf16* bp2 = Bp + ((size_t)((ng * 4 + 2) * KT + tk) * 32 + lane) * 16;
        const __bf16* bp3 = Bp + ((size_t)((ng * 4 + 3) * KT + tk) * 32 + lane) * 16;
        v16bf b0 = *(const v16bf*)bp0;
        v16bf b1 = *(const v16bf*)bp1;
        v16bf b2 = *(const v16bf*)bp2;
        v16bf b3 = *(const v16bf*)bp3;
        acc0 = __builtin_amdgcn_wmma_f32_16x16x32_bf16(false, a, false, b0, (short)0, acc0, false, false);
        acc1 = __builtin_amdgcn_wmma_f32_16x16x32_bf16(false, a, false, b1, (short)0, acc1, false, false);
        acc2 = __builtin_amdgcn_wmma_f32_16x16x32_bf16(false, a, false, b2, (short)0, acc2, false, false);
        acc3 = __builtin_amdgcn_wmma_f32_16x16x32_bf16(false, a, false, b3, (short)0, acc3, false, false);
    }

    // epilogue: C/D layout  m = mtile*16 + (lane/16)*8 + v,  n = ntile*16 + lane%16
    v8f accs[4] = {acc0, acc1, acc2, acc3};
#pragma unroll
    for (int j = 0; j < 4; ++j) {
        int n = (ng * 4 + j) * 16 + (lane & 15);
        float bv = bias ? bias[n] : 0.0f;
#pragma unroll
        for (int v = 0; v < 8; ++v) {
            int mm = mtile * 16 + ((lane >> 4) << 3) + v;
            float r = accs[j][v] + bv;
            if (act == 1) r = fmaxf(r, 0.0f);
            Y[(size_t)mm * N + n] = r;
        }
    }
}

// ============================================================
// Fused GRU: one wave owns a 16(M) x 16(N) h_new tile.
// Accumulates 6 gate tiles (ir,iz,in from x@Wih; hr,hz,hn from h@Whh),
// applies gate math in the epilogue, writes h_new. Removes the gi/gh
// round-trip (3 MB/step) and 2 launches from the recurrence.
// ============================================================
__global__ __launch_bounds__(256)
void gru_fused_kernel(const float* __restrict__ x, const float* __restrict__ h,
                      const __bf16* __restrict__ pWih, const __bf16* __restrict__ pWhh,
                      const float* __restrict__ bih, const float* __restrict__ bhh,
                      float* __restrict__ hnew) {
    const int NT = DHH >> 4;                       // 32 n-tiles over H
    const int KT = DHH >> 5;                       // 16 k-steps (K = H = 512)
    int wave = blockIdx.x * (blockDim.x >> 5) + (threadIdx.x >> 5);
    int lane = threadIdx.x & 31;
    if (wave >= (DB >> 4) * NT) return;            // wave-uniform
    int mtile = wave / NT;
    int nt    = wave % NT;

    v8f air = {}, aiz = {}, ain = {};
    v8f ahr = {}, ahz = {}, ahn = {};

    int m = mtile * 16 + (lane & 15);
    int khalf = (lane >> 4) << 3;
    const float* xrow = x + (size_t)m * DHH;
    const float* hrow = h + (size_t)m * DHH;

    for (int tk = 0; tk < KT; ++tk) {
        v16bf ax = build_a_frag(xrow, DHH, (const float*)0, tk * 32, khalf);
        v16bf ah = build_a_frag(hrow, DHH, (const float*)0, tk * 32, khalf);
        // gate g tile column index: tn = g*NT + nt  (Wih/Whh rows ordered r,z,n)
        const __bf16* bir = pWih + ((size_t)((0 * NT + nt) * KT + tk) * 32 + lane) * 16;
        const __bf16* biz = pWih + ((size_t)((1 * NT + nt) * KT + tk) * 32 + lane) * 16;
        const __bf16* bin = pWih + ((size_t)((2 * NT + nt) * KT + tk) * 32 + lane) * 16;
        const __bf16* bhr = pWhh + ((size_t)((0 * NT + nt) * KT + tk) * 32 + lane) * 16;
        const __bf16* bhz = pWhh + ((size_t)((1 * NT + nt) * KT + tk) * 32 + lane) * 16;
        const __bf16* bhn = pWhh + ((size_t)((2 * NT + nt) * KT + tk) * 32 + lane) * 16;
        v16bf t0 = *(const v16bf*)bir;
        v16bf t1 = *(const v16bf*)biz;
        v16bf t2 = *(const v16bf*)bin;
        v16bf t3 = *(const v16bf*)bhr;
        v16bf t4 = *(const v16bf*)bhz;
        v16bf t5 = *(const v16bf*)bhn;
        air = __builtin_amdgcn_wmma_f32_16x16x32_bf16(false, ax, false, t0, (short)0, air, false, false);
        aiz = __builtin_amdgcn_wmma_f32_16x16x32_bf16(false, ax, false, t1, (short)0, aiz, false, false);
        ain = __builtin_amdgcn_wmma_f32_16x16x32_bf16(false, ax, false, t2, (short)0, ain, false, false);
        ahr = __builtin_amdgcn_wmma_f32_16x16x32_bf16(false, ah, false, t3, (short)0, ahr, false, false);
        ahz = __builtin_amdgcn_wmma_f32_16x16x32_bf16(false, ah, false, t4, (short)0, ahz, false, false);
        ahn = __builtin_amdgcn_wmma_f32_16x16x32_bf16(false, ah, false, t5, (short)0, ahn, false, false);
    }

    int n  = nt * 16 + (lane & 15);
    float b_ir = bih[n], b_iz = bih[DHH + n], b_in = bih[2 * DHH + n];
    float b_hr = bhh[n], b_hz = bhh[DHH + n], b_hn = bhh[2 * DHH + n];
#pragma unroll
    for (int v = 0; v < 8; ++v) {
        int mm = mtile * 16 + ((lane >> 4) << 3) + v;
        float ir = air[v] + b_ir, iz = aiz[v] + b_iz, in = ain[v] + b_in;
        float hr = ahr[v] + b_hr, hz = ahz[v] + b_hz, hn = ahn[v] + b_hn;
        float r  = sigmoidf_fast(ir + hr);
        float z  = sigmoidf_fast(iz + hz);
        float nn = tanhf(in + r * hn);
        float hv = h[(size_t)mm * DHH + n];
        hnew[(size_t)mm * DHH + n] = (1.0f - z) * nn + z * hv;
    }
}

// ============================================================
// Per-step small kernels
// ============================================================

__global__ void embed_kernel(const int* __restrict__ ids, const float* __restrict__ emb,
                             float* __restrict__ e) {
    int b = blockIdx.x, t = threadIdx.x;               // E = 256 threads
    e[b * DE + t] = emb[(size_t)ids[b] * DE + t];
}

__global__ void softmax256_kernel(const float* __restrict__ S, float* __restrict__ P) {
    __shared__ float sm[256];
    int b = blockIdx.x, t = threadIdx.x;
    float v = S[b * 256 + t];
    sm[t] = v; __syncthreads();
    for (int o = 128; o > 0; o >>= 1) { if (t < o) sm[t] = fmaxf(sm[t], sm[t + o]); __syncthreads(); }
    float mx = sm[0]; __syncthreads();
    float ev = __expf(v - mx);
    sm[t] = ev; __syncthreads();
    for (int o = 128; o > 0; o >>= 1) { if (t < o) sm[t] += sm[t + o]; __syncthreads(); }
    P[b * 256 + t] = ev * (1.0f / sm[0]);
}

// ctx[b,h] = sum_t aw[b,t] * enc_bf16[b,t,h]   (bandwidth-bound, L2-resident enc)
__global__ void ctx_kernel(const float* __restrict__ aw, const __bf16* __restrict__ encb,
                           float* __restrict__ ctx) {
    __shared__ float a_s[TENC];
    int b = blockIdx.x, t = threadIdx.x;               // 256 threads, 4 cols each
    a_s[t] = aw[b * TENC + t];
    __syncthreads();
    float acc0 = 0.f, acc1 = 0.f, acc2 = 0.f, acc3 = 0.f;
    const __bf16* eb = encb + (size_t)b * TENC * D2H + t * 4;
    for (int k = 0; k < TENC; ++k) {
        float w = a_s[k];
        v4bf p = *(const v4bf*)(eb + (size_t)k * D2H);
        acc0 += w * (float)p[0];
        acc1 += w * (float)p[1];
        acc2 += w * (float)p[2];
        acc3 += w * (float)p[3];
    }
    float* cb = ctx + (size_t)b * D2H + t * 4;
    cb[0] = acc0; cb[1] = acc1; cb[2] = acc2; cb[3] = acc3;
}

// ============================================================
// CRF kernels
// ============================================================

__global__ void numerator_kernel(const float* __restrict__ em, const int* __restrict__ tags,
                                 const float* __restrict__ start, const float* __restrict__ endv,
                                 const float* __restrict__ trans, float* __restrict__ num) {
    __shared__ float sm[256];
    int b = blockIdx.x, t = threadIdx.x;               // T = 256 threads
    int tg = tags[t * DB + b];
    float v = em[((size_t)t * DB + b) * DK + tg];
    if (t < DT - 1) v += trans[tg * DK + tags[(t + 1) * DB + b]];
    if (t == 0)      v += start[tg];
    if (t == DT - 1) v += endv[tg];
    sm[t] = v; __syncthreads();
    for (int o = 128; o > 0; o >>= 1) { if (t < o) sm[t] += sm[t + o]; __syncthreads(); }
    if (t == 0) num[b] = sm[0];
}

// One block per batch element: sequential scan over T, score in LDS,
// trans streamed from L2 (256 KB, shared by all 128 blocks), online logsumexp.
__global__ __launch_bounds__(256)
void crf_forward_kernel(const float* __restrict__ em, const float* __restrict__ trans,
                        const float* __restrict__ start, const float* __restrict__ endv,
                        const float* __restrict__ num, float* __restrict__ llh) {
    __shared__ float sc[DK];
    __shared__ float red[DK];
    int b = blockIdx.x, j = threadIdx.x;               // K = 256 threads
    sc[j] = start[j] + em[(size_t)b * DK + j];
    __syncthreads();
    for (int t = 1; t < DT; ++t) {
        float m = -1e30f, s = 0.0f;
        const float* tr = trans + j;
        for (int i = 0; i < DK; ++i) {
            float v = sc[i] + tr[i * DK];              // LDS broadcast + coalesced global
            if (v > m) { s = s * __expf(m - v) + 1.0f; m = v; }
            else       { s += __expf(v - m); }
        }
        float ns = em[((size_t)t * DB + b) * DK + j] + m + __logf(s);
        __syncthreads();
        sc[j] = ns;
        __syncthreads();
    }
    float v = sc[j] + endv[j];
    red[j] = v; __syncthreads();
    for (int o = 128; o > 0; o >>= 1) { if (j < o) red[j] = fmaxf(red[j], red[j + o]); __syncthreads(); }
    float mx = red[0]; __syncthreads();
    red[j] = __expf(v - mx); __syncthreads();
    for (int o = 128; o > 0; o >>= 1) { if (j < o) red[j] += red[j + o]; __syncthreads(); }
    if (j == 0) llh[b] = num[b] - (mx + __logf(red[0]));
}

__global__ void mean_kernel(const float* __restrict__ llh, float* __restrict__ out) {
    __shared__ float sm[DB];
    int t = threadIdx.x;
    sm[t] = llh[t]; __syncthreads();
    for (int o = 64; o > 0; o >>= 1) { if (t < o) sm[t] += sm[t + o]; __syncthreads(); }
    if (t == 0) out[0] = sm[0] * (1.0f / DB);
}

// ============================================================
// Host: launch graph-capturable kernel DAG
// ============================================================
static inline size_t alignup(size_t x) { return (x + 255) & ~(size_t)255; }

static inline void launch_gemm(const float* X0, int K0, const float* X1,
                               const __bf16* Bp, const float* bias, float* Y,
                               int M, int N, int Ktot, int act, hipStream_t s) {
    int waves = (M >> 4) * (N >> 6);
    int blocks = (waves + 7) / 8;                      // 8 waves (256 threads) / block
    gemm_bf16_kernel<<<blocks, 256, 0, s>>>(X0, K0, X1, Bp, bias, Y, M, N, Ktot, act);
}

extern "C" void kernel_launch(void* const* d_in, const int* in_sizes, int n_in,
                              void* d_out, int out_size, void* d_ws, size_t ws_size,
                              hipStream_t stream) {
    (void)in_sizes; (void)n_in; (void)out_size; (void)ws_size;
    const int*   dec_input = (const int*)  d_in[0];    // (1,B,1)
    const int*   out_tags  = (const int*)  d_in[1];    // (T,B)
    const float* enc       = (const float*)d_in[2];    // (T,B,2H)
    const float* emb       = (const float*)d_in[3];    // (K,E)
    const float* attn_W    = (const float*)d_in[4];    // (256,768)
    const float* attn_b    = (const float*)d_in[5];
    const float* comb_W    = (const float*)d_in[6];    // (512,1280)
    const float* comb_b    = (const float*)d_in[7];
    const float* gru_Wih   = (const float*)d_in[8];    // (1536,512)
    const float* gru_Whh   = (const float*)d_in[9];    // (1536,512)
    const float* gru_bih   = (const float*)d_in[10];
    const float* gru_bhh   = (const float*)d_in[11];
    const float* out_W     = (const float*)d_in[12];   // (256,512)
    const float* out_b     = (const float*)d_in[13];
    const float* crf_start = (const float*)d_in[14];
    const float* crf_end   = (const float*)d_in[15];
    const float* crf_trans = (const float*)d_in[16];

    // ---- workspace carve ----
    char* ws = (char*)d_ws;
    size_t off = 0;
    __bf16* encb    = (__bf16*)(ws + off); off = alignup(off + (size_t)DB * TENC * D2H * 2);
    __bf16* pb_attn = (__bf16*)(ws + off); off = alignup(off + (size_t)768  * 256  * 2);
    __bf16* pb_comb = (__bf16*)(ws + off); off = alignup(off + (size_t)1280 * 512  * 2);
    __bf16* pb_wih  = (__bf16*)(ws + off); off = alignup(off + (size_t)512  * 1536 * 2);
    __bf16* pb_whh  = (__bf16*)(ws + off); off = alignup(off + (size_t)512  * 1536 * 2);
    __bf16* pb_out  = (__bf16*)(ws + off); off = alignup(off + (size_t)512  * 256  * 2);
    float* emis = (float*)(ws + off); off = alignup(off + (size_t)DT * DB * DK * 4);
    float* h0   = (float*)(ws + off); off = alignup(off + (size_t)DB * DHH * 4);
    float* h1   = (float*)(ws + off); off = alignup(off + (size_t)DB * DHH * 4);
    float* ebuf = (float*)(ws + off); off = alignup(off + (size_t)DB * DE * 4);
    float* sbuf = (float*)(ws + off); off = alignup(off + (size_t)DB * TENC * 4);
    float* awb  = (float*)(ws + off); off = alignup(off + (size_t)DB * TENC * 4);
    float* ctxb = (float*)(ws + off); off = alignup(off + (size_t)DB * D2H * 4);
    float* xbuf = (float*)(ws + off); off = alignup(off + (size_t)DB * DHH * 4);
    float* numb = (float*)(ws + off); off = alignup(off + (size_t)DB * 4);
    float* llhb = (float*)(ws + off); off = alignup(off + (size_t)DB * 4);

    // ---- prep: enc -> (B,T,2H) bf16 ; pack all weights to WMMA B fragments ----
    convert_enc_kernel<<<(size_t)DB * TENC * D2H / 256, 256, 0, stream>>>(enc, encb);
    pack_b_kernel<<<768  * 256  / 256, 256, 0, stream>>>(attn_W,  pb_attn, 256,  768);
    pack_b_kernel<<<1280 * 512  / 256, 256, 0, stream>>>(comb_W,  pb_comb, 512,  1280);
    pack_b_kernel<<<512  * 1536 / 256, 256, 0, stream>>>(gru_Wih, pb_wih,  1536, 512);
    pack_b_kernel<<<512  * 1536 / 256, 256, 0, stream>>>(gru_Whh, pb_whh,  1536, 512);
    pack_b_kernel<<<512  * 256  / 256, 256, 0, stream>>>(out_W,   pb_out,  256,  512);
    zero_kernel<<<(DB * DHH + 255) / 256, 256, 0, stream>>>(h0, DB * DHH);

    // ---- sequential decoder (teacher forcing); h ping-pongs h0 <-> h1 ----
    float* h_cur = h0;
    float* h_nxt = h1;
    for (int t = 0; t < DT; ++t) {
        const int* ids = (t == 0) ? dec_input : (out_tags + (size_t)(t - 1) * DB);
        embed_kernel<<<DB, 256, 0, stream>>>(ids, emb, ebuf);
        // scores = [e|h] @ attn_W^T + attn_b
        launch_gemm(ebuf, DE, h_cur, pb_attn, attn_b, sbuf, DB, TENC, DE + DHH, 0, stream);
        softmax256_kernel<<<DB, 256, 0, stream>>>(sbuf, awb);
        ctx_kernel<<<DB, 256, 0, stream>>>(awb, encb, ctxb);
        // x = relu([e|ctx] @ comb_W^T + comb_b)
        launch_gemm(ebuf, DE, ctxb, pb_comb, comb_b, xbuf, DB, DHH, DE + D2H, 1, stream);
        // fused GRU: h_nxt = GRU(x, h_cur)   (6 WMMA tiles per wave + gate epilogue)
        gru_fused_kernel<<<32, 256, 0, stream>>>(xbuf, h_cur, pb_wih, pb_whh,
                                                 gru_bih, gru_bhh, h_nxt);
        // emissions[t] = h_new @ out_W^T + out_b
        launch_gemm(h_nxt, DHH, (const float*)0, pb_out, out_b, emis + (size_t)t * DB * DK,
                    DB, DK, DHH, 0, stream);
        float* tmp = h_cur; h_cur = h_nxt; h_nxt = tmp;
    }

    // ---- CRF: numerator, forward scan (per-batch independent), mean ----
    numerator_kernel<<<DB, 256, 0, stream>>>(emis, out_tags, crf_start, crf_end, crf_trans, numb);
    crf_forward_kernel<<<DB, 256, 0, stream>>>(emis, crf_trans, crf_start, crf_end, numb, llhb);
    mean_kernel<<<1, DB, 0, stream>>>(llhb, (float*)d_out);
}
